// BitFeedForward_25280177504455
// MI455X (gfx1250) — 8-bit matrix operand order repaired and checked by reading, compiled, not run on hardware
//
#include <hip/hip_runtime.h>

typedef __attribute__((ext_vector_type(8))) int v8i;

#define DIM_   2048
#define INNER_ 8192
#define TOK_   8192

#define OFF_SUM  ((size_t)0)
#define OFF_W1Q  ((size_t)256)
#define OFF_W2Q  (OFF_W1Q + (size_t)INNER_ * DIM_)
#define OFF_XQ   (OFF_W2Q + (size_t)DIM_ * INNER_)
#define OFF_HQ   (OFF_XQ  + (size_t)TOK_ * DIM_)
#define OFF_F1   (OFF_HQ  + (size_t)TOK_ * INNER_)
#define OFF_F2   (OFF_F1  + (size_t)TOK_ * 4)
#define OFF_H    (OFF_F2  + (size_t)TOK_ * 4)

__global__ void init_sums_kernel(float* s) {
  if (threadIdx.x < 2) s[threadIdx.x] = 0.0f;
}

__global__ __launch_bounds__(256) void weight_quant_kernel(
    const float* __restrict__ w, signed char* __restrict__ wq, float* __restrict__ sum_out) {
  __shared__ float red[256];
  const int tid = threadIdx.x;
  const size_t idx = (size_t)blockIdx.x * 256 + tid;
  float4 v = reinterpret_cast<const float4*>(w)[idx];
  int q0 = (v.x > 0.0f) - (v.x < 0.0f);
  int q1 = (v.y > 0.0f) - (v.y < 0.0f);
  int q2 = (v.z > 0.0f) - (v.z < 0.0f);
  int q3 = (v.w > 0.0f) - (v.w < 0.0f);
  reinterpret_cast<int*>(wq)[idx] =
      (q0 & 255) | ((q1 & 255) << 8) | ((q2 & 255) << 16) | ((q3 & 255) << 24);
  red[tid] = fabsf(v.x) + fabsf(v.y) + fabsf(v.z) + fabsf(v.w);
  __syncthreads();
  for (int s = 128; s > 0; s >>= 1) {
    if (tid < s) red[tid] += red[tid + s];
    __syncthreads();
  }
  if (tid == 0) atomicAdd(sum_out, red[0]);
}

template<int W>
__global__ __launch_bounds__(256) void act_quant_kernel(
    const float* __restrict__ x, signed char* __restrict__ xq, float* __restrict__ f,
    const float* __restrict__ wsum, float wcnt_inv) {
  constexpr int NV = W / 1024;
  __shared__ float rsum[256], rmax[256];
  const int tid = threadIdx.x;
  const size_t base = (size_t)blockIdx.x * W;
  const float4* xv = reinterpret_cast<const float4*>(x + base);
  float4 vals[NV];
  float ss = 0.0f, mx = 0.0f;
#pragma unroll
  for (int i = 0; i < NV; ++i) {
    float4 v = xv[tid + i * 256];
    vals[i] = v;
    ss += v.x * v.x + v.y * v.y + v.z * v.z + v.w * v.w;
    mx = fmaxf(mx, fmaxf(fmaxf(fabsf(v.x), fabsf(v.y)), fmaxf(fabsf(v.z), fabsf(v.w))));
  }
  rsum[tid] = ss; rmax[tid] = mx;
  __syncthreads();
  for (int s = 128; s > 0; s >>= 1) {
    if (tid < s) {
      rsum[tid] += rsum[tid + s];
      rmax[tid] = fmaxf(rmax[tid], rmax[tid + s]);
    }
    __syncthreads();
  }
  const float inv = 1.0f / (sqrtf(rsum[0] * (1.0f / (float)W) + 1e-6f) * sqrtf((float)W));
  const float scale = 127.0f / fmaxf(rmax[0] * inv, 1e-5f);
  if (tid == 0) f[blockIdx.x] = (*wsum) * wcnt_inv / scale;
  const float a = inv * scale;
  int* out = reinterpret_cast<int*>(xq + base);
#pragma unroll
  for (int i = 0; i < NV; ++i) {
    float4 v = vals[i];
    int q0 = min(127, max(-128, (int)rintf(v.x * a)));
    int q1 = min(127, max(-128, (int)rintf(v.y * a)));
    int q2 = min(127, max(-128, (int)rintf(v.z * a)));
    int q3 = min(127, max(-128, (int)rintf(v.w * a)));
    out[tid + i * 256] = (q0 & 255) | ((q1 & 255) << 8) | ((q2 & 255) << 16) | ((q3 & 255) << 24);
  }
}

template<bool GELU>
__global__ __launch_bounds__(256) void gemm_iu8_kernel(
    const signed char* __restrict__ A, const signed char* __restrict__ B,
    const float* __restrict__ f, const float* __restrict__ bias,
    float* __restrict__ C, int N, int K) {
  __shared__ __align__(16) unsigned char lsA[256 * 144];
  __shared__ __align__(16) unsigned char lsB[128 * 144];
  const int tid  = threadIdx.x;
  const int lane = tid & 31;
  const int wave = tid >> 5;
  const int wm   = wave >> 1;
  const int wn   = wave & 1;
  const int r16  = lane & 15;
  const int half = lane >> 4;
  const size_t tileM = (size_t)blockIdx.y * 256;
  const size_t tileN = (size_t)blockIdx.x * 128;

  v8i acc[4][4];
#pragma unroll
  for (int i = 0; i < 4; ++i)
#pragma unroll
    for (int j = 0; j < 4; ++j) acc[i][j] = 0;

  const int srow = tid >> 1;
  const int scol = (tid & 1) * 64;
  const signed char* gA0 = A + (tileM + srow)       * (size_t)K + scol;
  const signed char* gA1 = A + (tileM + srow + 128) * (size_t)K + scol;
  const signed char* gB  = B + (tileN + srow)       * (size_t)K + scol;

  for (int kb = 0; kb < K; kb += 128) {
    if (kb + 128 < K) {
      __builtin_prefetch(gA0 + kb + 128, 0, 1);
      __builtin_prefetch(gB  + kb + 128, 0, 1);
    }
#pragma unroll
    for (int c = 0; c < 4; ++c) {
      *reinterpret_cast<uint4*>(&lsA[srow * 144 + scol + c * 16]) =
          *reinterpret_cast<const uint4*>(gA0 + kb + c * 16);
      *reinterpret_cast<uint4*>(&lsA[(srow + 128) * 144 + scol + c * 16]) =
          *reinterpret_cast<const uint4*>(gA1 + kb + c * 16);
      *reinterpret_cast<uint4*>(&lsB[srow * 144 + scol + c * 16]) =
          *reinterpret_cast<const uint4*>(gB + kb + c * 16);
    }
    __syncthreads();

#pragma unroll
    for (int ks = 0; ks < 128; ks += 64) {
      v8i bf[4];
#pragma unroll
      for (int ni = 0; ni < 4; ++ni) {
        const unsigned char* rp = &lsB[(wn * 64 + ni * 16 + r16) * 144 + ks];
        uint4 p0 = *reinterpret_cast<const uint4*>(rp + half * 16);
        uint4 p1 = *reinterpret_cast<const uint4*>(rp + 32 + half * 16);
        v8i b;
        b[0] = (int)p0.x; b[1] = (int)p0.y; b[2] = (int)p0.z; b[3] = (int)p0.w;
        b[4] = (int)p1.x; b[5] = (int)p1.y; b[6] = (int)p1.z; b[7] = (int)p1.w;
        bf[ni] = b;
      }
#pragma unroll
      for (int mi = 0; mi < 4; ++mi) {
        const unsigned char* rp = &lsA[(wm * 64 + mi * 16 + r16) * 144 + ks];
        uint2 c0 = *reinterpret_cast<const uint2*>(rp +      half * 16);
        uint2 c1 = *reinterpret_cast<const uint2*>(rp + 8  + half * 16);
        uint2 c2 = *reinterpret_cast<const uint2*>(rp + 32 + half * 16);
        uint2 c3 = *reinterpret_cast<const uint2*>(rp + 40 + half * 16);
        v8i a;
        a[0] = (int)c0.x; a[1] = (int)c0.y; a[2] = (int)c1.x; a[3] = (int)c1.y;
        a[4] = (int)c2.x; a[5] = (int)c2.y; a[6] = (int)c3.x; a[7] = (int)c3.y;
#pragma unroll
        for (int ni = 0; ni < 4; ++ni) {
          acc[mi][ni] = __builtin_amdgcn_wmma_i32_16x16x64_iu8(
              true, a, true, bf[ni], acc[mi][ni], false, false);
        }
      }
    }
    __syncthreads();
  }

#pragma unroll
  for (int mi = 0; mi < 4; ++mi) {
    const size_t mbase = tileM + (size_t)wm * 64 + mi * 16 + half * 8;
    float fr[8];
#pragma unroll
    for (int j = 0; j < 8; ++j) fr[j] = f[mbase + j];
#pragma unroll
    for (int ni = 0; ni < 4; ++ni) {
      const size_t gcol = tileN + (size_t)wn * 64 + ni * 16 + r16;
      const float bc = bias[gcol];
#pragma unroll
      for (int j = 0; j < 8; ++j) {
        float v = (float)acc[mi][ni][j] * fr[j] + bc;
        if (GELU) v = 0.5f * v * (1.0f + erff(v * 0.70710678118654752f));
        C[(mbase + j) * (size_t)N + gcol] = v;
      }
    }
  }
}

extern "C" void kernel_launch(void* const* d_in, const int* in_sizes, int n_in,
                              void* d_out, int out_size, void* d_ws, size_t ws_size,
                              hipStream_t stream) {
  const float* x  = (const float*)d_in[0];
  const float* w1 = (const float*)d_in[1];
  const float* b1 = (const float*)d_in[2];
  const float* w2 = (const float*)d_in[3];
  const float* b2 = (const float*)d_in[4];

  char* ws = (char*)d_ws;
  float*       sums = (float*)(ws + OFF_SUM);
  signed char* w1q  = (signed char*)(ws + OFF_W1Q);
  signed char* w2q  = (signed char*)(ws + OFF_W2Q);
  signed char* xq   = (signed char*)(ws + OFF_XQ);
  signed char* hq   = (signed char*)(ws + OFF_HQ);
  float*       f1   = (float*)(ws + OFF_F1);
  float*       f2   = (float*)(ws + OFF_F2);
  float*       h    = (float*)(ws + OFF_H);
  float*       out  = (float*)d_out;

  const int wblocks = (int)(((size_t)INNER_ * DIM_ / 4) / 256);

  init_sums_kernel<<<1, 32, 0, stream>>>(sums);
  weight_quant_kernel<<<wblocks, 256, 0, stream>>>(w1, w1q, &sums[0]);
  weight_quant_kernel<<<wblocks, 256, 0, stream>>>(w2, w2q, &sums[1]);

  act_quant_kernel<DIM_><<<TOK_, 256, 0, stream>>>(
      x, xq, f1, &sums[0], 1.0f / ((float)INNER_ * (float)DIM_));

  gemm_iu8_kernel<true><<<dim3(INNER_ / 128, TOK_ / 256), 256, 0, stream>>>(
      xq, w1q, f1, b1, h, INNER_, DIM_);

  act_quant_kernel<INNER_><<<TOK_, 256, 0, stream>>>(
      h, hq, f2, &sums[1], 1.0f / ((float)DIM_ * (float)INNER_));

  gemm_iu8_kernel<false><<<dim3(DIM_ / 128, TOK_ / 256), 256, 0, stream>>>(
      hq, w2q, f2, b2, out, DIM_, INNER_);
}
